// MusicVAE_1769526526319
// MI455X (gfx1250) — compile-verified, matrix operand order checked by reading, not run on hardware
//
#include <hip/hip_runtime.h>
#include <hip/hip_bf16.h>

// ---------------------------------------------------------------------------
// MusicVAE inference for gfx1250 (MI455X).  All heavy math goes through a
// bf16 WMMA GEMM (v_wmma_f32_16x16x32_bf16, wave32).  Weight tiles are staged
// into LDS with CDNA5 async-to-LDS loads (ASYNCcnt), double-buffered so the
// next tile's fetch overlaps the current tile's 4 back-to-back WMMAs.
// ---------------------------------------------------------------------------

typedef __bf16 bf16;
typedef __attribute__((ext_vector_type(16))) __bf16 v16bf;
typedef __attribute__((ext_vector_type(8)))  __bf16 v8bf;
typedef __attribute__((ext_vector_type(8)))  float  v8f;
typedef __attribute__((ext_vector_type(4)))  int    v4i;

#define FLAG_ADDC 1
#define FLAG_TANH 2

#if defined(__AMDGCN__) && \
    __has_builtin(__builtin_amdgcn_global_load_async_to_lds_b128) && \
    __has_builtin(__builtin_amdgcn_s_wait_asynccnt)
#define USE_ASYNC_LDS 1
typedef __attribute__((address_space(1))) v4i g_v4i;   // global int4
typedef __attribute__((address_space(3))) v4i l_v4i;   // LDS int4
__device__ __forceinline__ g_v4i* to_global_v4i(const void* p) {
  return (g_v4i*)(unsigned long long)p;                // AS1 == flat address
}
__device__ __forceinline__ l_v4i* to_lds_v4i(const void* p) {
  return (l_v4i*)(unsigned)(unsigned long long)p;      // AS3 == low 32 bits
}
#else
#define USE_ASYNC_LDS 0
#endif

__device__ __forceinline__ bf16 f2bf(float f) {
  union { float f; unsigned u; } x; x.f = f;
  unsigned r = x.u + 0x7FFFu + ((x.u >> 16) & 1u);   // round-to-nearest-even
  unsigned short h = (unsigned short)(r >> 16);
  return __builtin_bit_cast(bf16, h);
}
__device__ __forceinline__ float sigf(float x) { return 1.f / (1.f + __expf(-x)); }

// one K-chunk of a 16x64 wave tile: load A frag + 4 B frags, 4 WMMAs in a burst
__device__ __forceinline__ void mma_step(const bf16* __restrict__ Arow,
                                         const bf16* __restrict__ sBc, // [64][32]
                                         int ll, int lh, v8f acc[4]) {
  v8bf a0 = *(const v8bf*)(Arow);
  v8bf a1 = *(const v8bf*)(Arow + 16);
  v16bf af;
  #pragma unroll
  for (int i = 0; i < 8; ++i) { af[i] = a0[i]; af[i + 8] = a1[i]; }
  v16bf bfr[4];
  #pragma unroll
  for (int nt = 0; nt < 4; ++nt) {
    const bf16* Bp = sBc + (nt * 16 + ll) * 32 + lh * 8;
    v8bf b0 = *(const v8bf*)(Bp);
    v8bf b1 = *(const v8bf*)(Bp + 16);
    #pragma unroll
    for (int i = 0; i < 8; ++i) { bfr[nt][i] = b0[i]; bfr[nt][i + 8] = b1[i]; }
  }
  #pragma unroll
  for (int nt = 0; nt < 4; ++nt)
    acc[nt] = __builtin_amdgcn_wmma_f32_16x16x32_bf16(
        false, af, false, bfr[nt], (short)0, acc[nt], false, false);
}

// ---------------------------------------------------------------------------
// Generic WMMA GEMM:  C[M,N] (f32, optional) = act( A[M,K]bf16 @ Bt[N,K]bf16^T
//                     + bias + (addC ? Cin : 0) ),  optional bf16 copy Cb.
// Block: 256 threads = 8 waves.  Block tile 128x64, wave tile 16x64, K step 32.
// M multiple of 128; Bt allocated with ceil(N/64)*64 zero-padded rows; K a
// multiple of 32 (zero padded).
// ---------------------------------------------------------------------------
__global__ __launch_bounds__(256) void wmma_gemm_k(
    const bf16* __restrict__ A, long lda,
    const bf16* __restrict__ Bt, long ldb,
    const float* __restrict__ bias,
    float* __restrict__ C, long ldc,
    bf16* __restrict__ Cb, long ldcb,
    int M, int N, int K, int flags)
{
  __shared__ bf16 sB[2][64][32];              // 2 x 4KB weight tiles

  const int tid  = threadIdx.x;
  const int wave = tid >> 5;
  const int lane = tid & 31;
  const int lh   = lane >> 4;                 // lane half selects K sub-block
  const int ll   = lane & 15;
  const int mBase = blockIdx.x * 128 + wave * 16;
  const int nBase = blockIdx.y * 64;
  const int ln  = tid >> 2;                   // B tile row this thread stages
  const int lkc = (tid & 3) * 8;              // B tile k offset (x8 bf16 = 16B)

  v8f acc[4];
  #pragma unroll
  for (int i = 0; i < 4; ++i) acc[i] = v8f{0.f,0.f,0.f,0.f,0.f,0.f,0.f,0.f};

  const bf16* Abase = A + (long)(mBase + ll) * lda + lh * 8;
  const int nk = K >> 5;

#if USE_ASYNC_LDS
  // ---- async double-buffered pipeline ----
  {
    const bf16* gp = Bt + (long)(nBase + ln) * ldb + lkc;
    __builtin_amdgcn_global_load_async_to_lds_b128(
        to_global_v4i(gp), to_lds_v4i(&sB[0][ln][lkc]), 0, 0);
  }
  for (int i = 0; i < nk; ++i) {
    __builtin_amdgcn_s_wait_asynccnt(0);      // tile i resident in LDS
    __syncthreads();                          // ...for every wave
    if (i + 1 < nk) {                         // prefetch tile i+1 during compute
      const bf16* gp = Bt + (long)(nBase + ln) * ldb + (i + 1) * 32 + lkc;
      __builtin_amdgcn_global_load_async_to_lds_b128(
          to_global_v4i(gp), to_lds_v4i(&sB[(i + 1) & 1][ln][lkc]), 0, 0);
    }
    mma_step(Abase + (long)i * 32, &sB[i & 1][0][0], ll, lh, acc);
  }
#else
  // ---- fallback: synchronous staging through VGPRs ----
  for (int i = 0; i < nk; ++i) {
    v8bf bv = *(const v8bf*)(Bt + (long)(nBase + ln) * ldb + i * 32 + lkc);
    *(v8bf*)(&sB[0][ln][lkc]) = bv;
    __syncthreads();
    mma_step(Abase + (long)i * 32, &sB[0][0][0], ll, lh, acc);
    __syncthreads();
  }
#endif

  // epilogue: bias / add-C / tanh / f32 store / bf16 store
  #pragma unroll
  for (int nt = 0; nt < 4; ++nt) {
    const int n = nBase + nt * 16 + ll;
    if (n >= N) continue;
    const float bv = bias ? bias[n] : 0.f;
    #pragma unroll
    for (int r = 0; r < 8; ++r) {
      const int m = mBase + r + lh * 8;
      float v = acc[nt][r] + bv;
      if (flags & FLAG_ADDC) v += C[(long)m * ldc + n];
      if (flags & FLAG_TANH) v = tanhf(v);
      if (C)  C[(long)m * ldc + n]   = v;
      if (Cb) Cb[(long)m * ldcb + n] = f2bf(v);
    }
  }
}

// ---------------------------------------------------------------------------
// Elementwise kernels
// ---------------------------------------------------------------------------
__global__ void fill_u32_k(unsigned* p, long n) {
  long i = (long)blockIdx.x * blockDim.x + threadIdx.x;
  if (i < n) p[i] = 0u;
}

// f32 -> bf16 with zero padding; dst region is [drows, dcols] at column dcol0
__global__ void cvt_pad_k(const float* __restrict__ src, int srows, int scols,
                          bf16* __restrict__ dst, long dld, long dcol0,
                          int drows, int dcols) {
  long i = (long)blockIdx.x * blockDim.x + threadIdx.x;
  long tot = (long)drows * dcols;
  if (i >= tot) return;
  int r = (int)(i / dcols), c = (int)(i % dcols);
  float v = (r < srows && c < scols) ? src[(long)r * scols + c] : 0.f;
  dst[(long)r * dld + dcol0 + c] = f2bf(v);
}

// x[B,T,129] -> bf16 [T*B, 160] (time-major, K padded with zeros)
__global__ void cvt_x_k(const float* __restrict__ x, bf16* __restrict__ dst) {
  long i = (long)blockIdx.x * blockDim.x + threadIdx.x;
  const long tot = 32L * 1024 * 160;
  if (i >= tot) return;
  int c = (int)(i % 160);
  long row = i / 160;
  int b = (int)(row % 1024);
  int t = (int)(row / 1024);
  float v = (c < 129) ? x[((long)b * 32 + t) * 129 + c] : 0.f;
  dst[i] = f2bf(v);
}

// fused LSTM gate: gates[B,4H] (i,f,g,o) -> c update, h = sig(o)*tanh(c) bf16
__global__ void lstm_gate_k(const float* __restrict__ g, float* __restrict__ c,
                            bf16* __restrict__ h0, long h0ld,
                            bf16* __restrict__ h1, long h1ld, int H, long total) {
  long t = (long)blockIdx.x * blockDim.x + threadIdx.x;
  if (t >= total) return;
  int b = (int)(t / H), j = (int)(t % H);
  const float* gr = g + (long)b * 4 * H;
  float gi = sigf(gr[j]);
  float gf = sigf(gr[H + j]);
  float gg = tanhf(gr[2 * H + j]);
  float go = sigf(gr[3 * H + j]);
  float cc = gf * c[t] + gi * gg;
  c[t] = cc;
  bf16 hb = f2bf(go * tanhf(cc));
  if (h0) h0[(long)b * h0ld + j] = hb;
  if (h1) h1[(long)b * h1ld + j] = hb;
}

// z = eps * softplus(sig) + mu  -> bf16 [B,128]
__global__ void reparam_k(const float* __restrict__ ms, const float* __restrict__ eps,
                          bf16* __restrict__ z) {
  long t = (long)blockIdx.x * blockDim.x + threadIdx.x;
  if (t >= 1024L * 128) return;
  int b = (int)(t / 128), j = (int)(t % 128);
  float mu = ms[(long)b * 256 + j];
  float sg = ms[(long)b * 256 + 128 + j];
  float sp = log1pf(__expf(sg));
  z[t] = f2bf(eps[t] * sp + mu);
}

__global__ void copy_bf16_k(const bf16* __restrict__ src, long sld,
                            bf16* __restrict__ dst, long dld, int rows, int cols) {
  long t = (long)blockIdx.x * blockDim.x + threadIdx.x;
  if (t >= (long)rows * cols) return;
  int r = (int)(t / cols), c = (int)(t % cols);
  dst[(long)r * dld + c] = src[(long)r * sld + c];
}

// per-row argmax over 129 logits (one wave32 per batch row), then gather
// embedding column: emb[j] = emb_W[j, idx] + emb_b[j] -> xin cols 128..191
__global__ __launch_bounds__(128) void argmax_embed_k(
    const float* __restrict__ logits, long lld,
    const float* __restrict__ emb_W, const float* __restrict__ emb_b,
    bf16* __restrict__ xin) {
  const int wave = threadIdx.x >> 5, lane = threadIdx.x & 31;
  const int b = blockIdx.x * 4 + wave;
  const float* lg = logits + (long)b * lld;
  float best = -3.4e38f; int bi = 0;
  for (int j = lane; j < 129; j += 32) {
    float v = lg[j];
    if (v > best) { best = v; bi = j; }
  }
  #pragma unroll
  for (int off = 16; off > 0; off >>= 1) {
    float ov = __shfl_down(best, off, 32);
    int   oi = __shfl_down(bi,   off, 32);
    if (ov > best || (ov == best && oi < bi)) { best = ov; bi = oi; }
  }
  bi = __shfl(bi, 0, 32);
  for (int j = lane; j < 64; j += 32) {
    float v = emb_W[(long)j * 129 + bi] + emb_b[j];
    xin[(long)b * 320 + 128 + j] = f2bf(v);
  }
}

// ---------------------------------------------------------------------------
// Host orchestration
// ---------------------------------------------------------------------------
extern "C" void kernel_launch(void* const* d_in, const int* in_sizes, int n_in,
                              void* d_out, int out_size, void* d_ws, size_t ws_size,
                              hipStream_t stream) {
  (void)in_sizes; (void)n_in; (void)out_size; (void)ws_size;
  const float* x     = (const float*)d_in[0];
  const float* eps   = (const float*)d_in[1];
  const float* emb_W = (const float*)d_in[2];
  const float* emb_b = (const float*)d_in[3];
  const float* eWih0 = (const float*)d_in[4];
  const float* eWhh0 = (const float*)d_in[5];
  const float* eb0   = (const float*)d_in[6];
  const float* eWih1 = (const float*)d_in[7];
  const float* eWhh1 = (const float*)d_in[8];
  const float* eb1   = (const float*)d_in[9];
  const float* linW  = (const float*)d_in[10];
  const float* linb  = (const float*)d_in[11];
  const float* l2cW  = (const float*)d_in[12];
  const float* l2cb  = (const float*)d_in[13];
  /* d_in[14] cond0_Wih: unused (cond layer0 input is all zeros) */
  const float* c0Whh = (const float*)d_in[15];
  const float* c0b   = (const float*)d_in[16];
  const float* c0Whr = (const float*)d_in[17];
  const float* c1Wih = (const float*)d_in[18];
  const float* c1Whh = (const float*)d_in[19];
  const float* c1b   = (const float*)d_in[20];
  const float* c1Whr = (const float*)d_in[21];
  const float* c2dW  = (const float*)d_in[22];
  const float* c2db  = (const float*)d_in[23];
  const float* dWih  = (const float*)d_in[24];
  const float* dWhh  = (const float*)d_in[25];
  const float* dbias = (const float*)d_in[26];
  const float* dWhr  = (const float*)d_in[27];
  const float* d2lW  = (const float*)d_in[28];
  const float* d2lb  = (const float*)d_in[29];
  float* out = (float*)d_out;

  // ---- workspace bump allocator ----
  char* wptr = (char*)d_ws;
  auto alloc = [&](size_t bytes) -> char* {
    char* p = wptr; wptr += (bytes + 255) & ~(size_t)255; return p;
  };
  // activations (bf16)
  bf16* xtb   = (bf16*)alloc(32768L * 160 * 2);      // [T*B, 160]
  bf16* ebf   = (bf16*)alloc(32768L * 64  * 2);      // [T*B, 64]
  bf16* h1s   = (bf16*)alloc(32L * 1024 * 1024 * 2); // [T,B,1024] (fwd|bwd)
  bf16* h2s   = (bf16*)alloc(1024L * 32768 * 2);     // [B, T*1024]
  bf16* hcur  = (bf16*)alloc(1024L * 512 * 2);
  bf16* hpre  = (bf16*)alloc(1024L * 256 * 2);
  bf16* zbf   = (bf16*)alloc(1024L * 128 * 2);
  bf16* zcbf  = (bf16*)alloc(1024L * 256 * 2);
  bf16* y0b   = (bf16*)alloc(2L * 1024 * 128 * 2);
  bf16* dech  = (bf16*)alloc(2L * 1024 * 128 * 2);
  bf16* xin1  = (bf16*)alloc(1024L * 256 * 2);
  bf16* xin   = (bf16*)alloc(1024L * 320 * 2);       // [dz | emb | h]
  // f32 state
  float* gates = (float*)alloc(1024L * 2048 * 4);
  float* ccur  = (float*)alloc(1024L * 512 * 4);
  float* ms    = (float*)alloc(1024L * 256 * 4);
  float* ccnd0 = (float*)alloc(1024L * 256 * 4);
  float* ccnd1 = (float*)alloc(1024L * 256 * 4);
  float* cdec  = (float*)alloc(1024L * 256 * 4);
  // bf16 weights [Npad, Kpad]
  bf16* wEmb  = (bf16*)alloc(64L   * 160 * 2);
  bf16* wIh0  = (bf16*)alloc(2L * 2048 * 64  * 2);
  bf16* wHh0  = (bf16*)alloc(2L * 2048 * 512 * 2);
  bf16* wIh1  = (bf16*)alloc(2L * 2048 * 1024 * 2);
  bf16* wHh1  = (bf16*)alloc(2L * 2048 * 512 * 2);
  bf16* wLin  = (bf16*)alloc(256L * 32768 * 2);
  bf16* wL2c  = (bf16*)alloc(256L * 128 * 2);
  bf16* wC0hh = (bf16*)alloc(1024L * 128 * 2);
  bf16* wC0hr = (bf16*)alloc(128L * 256 * 2);
  bf16* wC1   = (bf16*)alloc(1024L * 256 * 2);       // [Wih|Whh]
  bf16* wC1hr = (bf16*)alloc(128L * 256 * 2);
  bf16* wC2d  = (bf16*)alloc(128L * 128 * 2);
  bf16* wDec  = (bf16*)alloc(1024L * 320 * 2);       // [Wih(192)|Whh(128)]
  bf16* wDhr  = (bf16*)alloc(128L * 256 * 2);
  bf16* wD2l  = (bf16*)alloc(192L * 128 * 2);        // N padded 129->192

  // ---- launch helpers ----
  auto gemm = [&](const bf16* A, long lda, const bf16* Bt, long ldb,
                  const float* bias, float* C, long ldc, bf16* Cb, long ldcb,
                  int M, int N, int K, int flags) {
    dim3 g((unsigned)(M / 128), (unsigned)((N + 63) / 64));
    wmma_gemm_k<<<g, 256, 0, stream>>>(A, lda, Bt, ldb, bias, C, ldc, Cb, ldcb,
                                       M, N, K, flags);
  };
  auto cvt = [&](const float* s, int sr, int sc, bf16* dst, long dld, long dc0,
                 int dr, int dc) {
    long tot = (long)dr * dc;
    cvt_pad_k<<<(unsigned)((tot + 255) / 256), 256, 0, stream>>>(
        s, sr, sc, dst, dld, dc0, dr, dc);
  };
  auto fillz = [&](void* p, size_t bytes) {
    long n = (long)(bytes / 4);
    fill_u32_k<<<(unsigned)((n + 255) / 256), 256, 0, stream>>>((unsigned*)p, n);
  };
  auto gate = [&](float* g, float* c, bf16* h0, long h0ld, bf16* h1, long h1ld,
                  int H) {
    long tot = 1024L * H;
    lstm_gate_k<<<(unsigned)((tot + 255) / 256), 256, 0, stream>>>(
        g, c, h0, h0ld, h1, h1ld, H, tot);
  };
  auto cp = [&](const bf16* s, long sld, bf16* d, long dld, int r, int c) {
    long tot = (long)r * c;
    copy_bf16_k<<<(unsigned)((tot + 255) / 256), 256, 0, stream>>>(s, sld, d, dld, r, c);
  };

  // ---- 1. weight / input conversion to padded bf16 ----
  cvt_x_k<<<(unsigned)((32768L * 160 + 255) / 256), 256, 0, stream>>>(x, xtb);
  cvt(emb_W, 64, 129, wEmb, 160, 0, 64, 160);
  for (int d = 0; d < 2; ++d) {
    cvt(eWih0 + (long)d * 2048 * 64,   2048, 64,   wIh0 + (long)d * 2048 * 64,   64,   0, 2048, 64);
    cvt(eWhh0 + (long)d * 2048 * 512,  2048, 512,  wHh0 + (long)d * 2048 * 512,  512,  0, 2048, 512);
    cvt(eWih1 + (long)d * 2048 * 1024, 2048, 1024, wIh1 + (long)d * 2048 * 1024, 1024, 0, 2048, 1024);
    cvt(eWhh1 + (long)d * 2048 * 512,  2048, 512,  wHh1 + (long)d * 2048 * 512,  512,  0, 2048, 512);
  }
  cvt(linW, 256, 32768, wLin, 32768, 0, 256, 32768);
  cvt(l2cW, 256, 128, wL2c, 128, 0, 256, 128);
  cvt(c0Whh, 1024, 128, wC0hh, 128, 0, 1024, 128);
  cvt(c0Whr, 128, 256, wC0hr, 256, 0, 128, 256);
  cvt(c1Wih, 1024, 128, wC1, 256, 0,   1024, 128);
  cvt(c1Whh, 1024, 128, wC1, 256, 128, 1024, 128);
  cvt(c1Whr, 128, 256, wC1hr, 256, 0, 128, 256);
  cvt(c2dW, 128, 128, wC2d, 128, 0, 128, 128);
  cvt(dWih, 1024, 192, wDec, 320, 0,   1024, 192);
  cvt(dWhh, 1024, 128, wDec, 320, 192, 1024, 128);
  cvt(dWhr, 128, 256, wDhr, 256, 0, 128, 256);
  cvt(d2lW, 129, 128, wD2l, 128, 0, 192, 128);       // pad rows 129..191 = 0

  // ---- 2. embedding: e = x @ emb_W^T + emb_b  (time-major) ----
  gemm(xtb, 160, wEmb, 160, emb_b, nullptr, 0, ebf, 64, 32768, 64, 160, 0);

  // ---- 3. encoder layer0, both directions ----
  for (int d = 0; d < 2; ++d) {
    fillz(ccur, 1024L * 512 * 4);
    fillz(hcur, 1024L * 512 * 2);
    for (int s = 0; s < 32; ++s) {
      int t = (d == 0) ? s : 31 - s;
      gemm(ebf + (long)t * 1024 * 64, 64, wIh0 + (long)d * 2048 * 64, 64,
           eb0 + (long)d * 2048, gates, 2048, nullptr, 0, 1024, 2048, 64, 0);
      gemm(hcur, 512, wHh0 + (long)d * 2048 * 512, 512, nullptr,
           gates, 2048, nullptr, 0, 1024, 2048, 512, FLAG_ADDC);
      gate(gates, ccur, hcur, 512,
           h1s + (long)t * 1024 * 1024 + (long)d * 512, 1024, 512);
    }
  }
  // ---- 4. encoder layer1, both directions ----
  for (int d = 0; d < 2; ++d) {
    fillz(ccur, 1024L * 512 * 4);
    fillz(hcur, 1024L * 512 * 2);
    for (int s = 0; s < 32; ++s) {
      int t = (d == 0) ? s : 31 - s;
      gemm(h1s + (long)t * 1024 * 1024, 1024, wIh1 + (long)d * 2048 * 1024, 1024,
           eb1 + (long)d * 2048, gates, 2048, nullptr, 0, 1024, 2048, 1024, 0);
      gemm(hcur, 512, wHh1 + (long)d * 2048 * 512, 512, nullptr,
           gates, 2048, nullptr, 0, 1024, 2048, 512, FLAG_ADDC);
      gate(gates, ccur, hcur, 512,
           h2s + (long)t * 1024 + (long)d * 512, 32768, 512);
    }
  }

  // ---- 5. enc_lin + reparam + latent_to_cond ----
  gemm(h2s, 32768, wLin, 32768, linb, ms, 256, nullptr, 0, 1024, 256, 32768, 0);
  reparam_k<<<(unsigned)((1024L * 128 + 255) / 256), 256, 0, stream>>>(ms, eps, zbf);
  gemm(zbf, 128, wL2c, 128, l2cb, nullptr, 0, zcbf, 256, 1024, 256, 128, FLAG_TANH);

  // ---- 6. cond LSTM layer0 (input is zeros; only Whh + bias) ----
  fillz(ccnd0, 1024L * 256 * 4);
  gemm(zcbf, 256, wC0hh, 128, c0b, gates, 1024, nullptr, 0, 1024, 1024, 128, 0);
  gate(gates, ccnd0, hpre, 256, nullptr, 0, 256);
  gemm(hpre, 256, wC0hr, 256, nullptr, nullptr, 0, y0b, 128, 1024, 128, 256, 0);
  gemm(y0b, 128, wC0hh, 128, c0b, gates, 1024, nullptr, 0, 1024, 1024, 128, 0);
  gate(gates, ccnd0, hpre, 256, nullptr, 0, 256);
  gemm(hpre, 256, wC0hr, 256, nullptr, nullptr, 0, y0b + 1024L * 128, 128,
       1024, 128, 256, 0);

  // ---- 7. cond LSTM layer1 ([y0_t | h] -> combined weights) ----
  fillz(ccnd1, 1024L * 256 * 4);
  cp(y0b, 128, xin1, 256, 1024, 128);
  cp(zcbf + 128, 256, xin1 + 128, 256, 1024, 128);   // h0 = zc[1]
  gemm(xin1, 256, wC1, 256, c1b, gates, 1024, nullptr, 0, 1024, 1024, 256, 0);
  gate(gates, ccnd1, hpre, 256, nullptr, 0, 256);
  gemm(hpre, 256, wC1hr, 256, nullptr, nullptr, 0, dech, 128, 1024, 128, 256, 0);
  cp(y0b + 1024L * 128, 128, xin1, 256, 1024, 128);
  cp(dech, 128, xin1 + 128, 256, 1024, 128);
  gemm(xin1, 256, wC1, 256, c1b, gates, 1024, nullptr, 0, 1024, 1024, 256, 0);
  gate(gates, ccnd1, hpre, 256, nullptr, 0, 256);
  gemm(hpre, 256, wC1hr, 256, nullptr, nullptr, 0, dech + 1024L * 128, 128,
       1024, 128, 256, 0);

  // ---- 8. decoder: 2 bars x 16 steps; cell persists, h resets to dz ----
  fillz(cdec, 1024L * 256 * 4);
  fillz(xin, 1024L * 320 * 2);                       // emb starts at zero
  for (int bar = 0; bar < 2; ++bar) {
    // dz = dec_h[bar] @ c2d^T + b  -> xin cols 0..127 (bf16)
    gemm(dech + (long)bar * 1024 * 128, 128, wC2d, 128, c2db,
         nullptr, 0, xin, 320, 1024, 128, 128, 0);
    cp(xin, 320, xin + 192, 320, 1024, 128);         // h = dz at bar start
    for (int s = 0; s < 16; ++s) {
      int tg = bar * 16 + s;
      // gates = [dz|emb|h] @ [Wih|Whh]^T + b
      gemm(xin, 320, wDec, 320, dbias, gates, 1024, nullptr, 0,
           1024, 1024, 320, 0);
      gate(gates, cdec, hpre, 256, nullptr, 0, 256);
      // h = h_pre @ Whr^T -> xin cols 192..319 (bf16)
      gemm(hpre, 256, wDhr, 256, nullptr, nullptr, 0, xin + 192, 320,
           1024, 128, 256, 0);
      // logits -> d_out[b, tg, :]  (ldc = 32*129)
      gemm(xin + 192, 320, wD2l, 128, d2lb, out + (long)tg * 129, 4128,
           nullptr, 0, 1024, 129, 128, 0);
      // argmax + embedding feedback -> xin cols 128..191
      argmax_embed_k<<<256, 128, 0, stream>>>(out + (long)tg * 129, 4128,
                                              emb_W, emb_b, xin);
    }
  }
}
